// CausalGDM_73023033967317
// MI455X (gfx1250) — matrix operand order repaired and checked by reading, not compiled, not run on hardware
//
#include <hip/hip_runtime.h>
#include <hip/hip_bf16.h>

// CausalGDM forward for MI455X (gfx1250, wave32, WMMA bf16 path).
// B=2, S=512, V=32000, D=512, FF=2048, 4 layers.

#define D_E 512
#define SEQ 512
#define BSZ 2
#define VOC 32000
#define DFF 2048
#define NLAYER 4
#define NSPLIT 4

typedef __attribute__((ext_vector_type(16))) __bf16 v16bf;
typedef __attribute__((ext_vector_type(8)))  float  v8f;

static __device__ inline v8f zero8() {
  v8f z = {0.f, 0.f, 0.f, 0.f, 0.f, 0.f, 0.f, 0.f};
  return z;
}

static __device__ inline v8f wmma_bf16(v16bf a, v16bf b, v8f c) {
  // args: (neg_a, A, neg_b, B, c_mod, C, reuse_a, reuse_b)
  return __builtin_amdgcn_wmma_f32_16x16x32_bf16(false, a, false, b, (short)0, c,
                                                 false, false);
}

// A fragment (16x32 bf16), row-major source with leading dim ld: A[r0+m][k0+k].
// Lanes 0-15: row M=lane, K 0..15; lanes 16-31: same rows, K 16..31.
static __device__ inline v16bf ld_A(const __bf16* A, int ld, int r0, int k0) {
  const int lane = threadIdx.x & 31;
  const __bf16* p = A + (long)(r0 + (lane & 15)) * ld + k0 + ((lane >> 4) << 4);
  v16bf a;
#pragma unroll
  for (int e = 0; e < 8; ++e) a[e] = p[e];
#pragma unroll
  for (int e = 0; e < 8; ++e) a[8 + e] = p[8 + e];
  return a;
}

// B fragment (32x16 bf16) from row-major (K x N): B[k][n] = M[(k0+k)*ld + n0+n].
static __device__ inline v16bf ld_B(const __bf16* M, int ld, int k0, int n0) {
  const int lane = threadIdx.x & 31;
  const __bf16* p = M + (long)(k0 + ((lane >> 4) << 4)) * ld + n0 + (lane & 15);
  v16bf b;
#pragma unroll
  for (int e = 0; e < 16; ++e) b[e] = p[(long)e * ld];
  return b;
}

// B fragment (32x16) when memory holds the transpose, row-major (N x K):
// B[k][n] = M[(n0+n)*ld + k0+k]  (contiguous 32B per lane).
static __device__ inline v16bf ld_BT(const __bf16* M, int ld, int k0, int n0) {
  const int lane = threadIdx.x & 31;
  const __bf16* p = M + (long)(n0 + (lane & 15)) * ld + k0 + ((lane >> 4) << 4);
  v16bf b;
#pragma unroll
  for (int e = 0; e < 16; ++e) b[e] = p[e];
  return b;
}

// ---------------- prep kernels ----------------

__global__ void k_cvt(const float* __restrict__ src, __bf16* __restrict__ dst, long n) {
  const long i = (long)blockIdx.x * 256 + threadIdx.x;
  if (i < n) dst[i] = (__bf16)src[i];
}

// tiled transpose: wteT[d*VOC + v] = bf16(wte[v*D + d]); 64x64 tiles via LDS
__global__ void k_transpose(const float* __restrict__ wte, __bf16* __restrict__ wteT) {
  __shared__ __bf16 tile[64][65];
  const int v0 = blockIdx.x * 64;
  const int d0 = blockIdx.y * 64;
  for (int idx = threadIdx.x; idx < 4096; idx += 256) {
    const int vv = idx >> 6, dd = idx & 63;
    tile[vv][dd] = (__bf16)wte[(long)(v0 + vv) * D_E + d0 + dd];
  }
  __syncthreads();
  for (int idx = threadIdx.x; idx < 4096; idx += 256) {
    const int dd = idx >> 6, vv = idx & 63;
    wteT[(long)(d0 + dd) * VOC + v0 + vv] = tile[vv][dd];
  }
}

// wosum[i,j] = sum_h w_o[i, h*512 + j]  (folds the N_HEAD tiling)
__global__ void k_wosum(const float* __restrict__ w_o, __bf16* __restrict__ wos_bf) {
  const long idx = (long)blockIdx.x * 256 + threadIdx.x;  // over D*D
  const long i = idx >> 9, j = idx & 511;
  float s = 0.f;
#pragma unroll
  for (int h = 0; h < 8; ++h) s += w_o[i * (long)(D_E * 8) + h * D_E + j];
  wos_bf[idx] = (__bf16)s;
}

__global__ void k_embed(const int* __restrict__ x, const float* __restrict__ wte,
                        float* __restrict__ e, float* __restrict__ fk,
                        __bf16* __restrict__ fkbf) {
  const long i = (long)blockIdx.x * 256 + threadIdx.x;  // over B*S*D
  const long bs = i >> 9;
  const int d = (int)(i & 511);
  const int tok = x[bs];
  e[i] = wte[(long)tok * D_E + d];
  fk[i] = 0.f;
  fkbf[i] = (__bf16)0.f;
}

static __device__ inline float block_red256(float v, float* red, bool ismax) {
#pragma unroll
  for (int off = 1; off < 32; off <<= 1) {
    float o = __shfl_xor(v, off, 32);
    v = ismax ? fmaxf(v, o) : v + o;
  }
  __syncthreads();  // protect red reuse across calls
  if ((threadIdx.x & 31) == 0) red[threadIdx.x >> 5] = v;
  __syncthreads();
  float r = red[0];
#pragma unroll
  for (int i = 1; i < 8; ++i) r = ismax ? fmaxf(r, red[i]) : r + red[i];
  return r;
}

// krn2[t, c] = softmax_c( clip(p[t+1]·p[c]/sqrt(D), ±10), mask c<=t ) / (t+1)
__global__ void k_krn(const float* __restrict__ wpe, __bf16* __restrict__ krn_bf) {
  __shared__ float prow[D_E];
  __shared__ float red[8];
  const int t = blockIdx.x;
  const int tid = threadIdx.x;
  for (int i = tid; i < D_E; i += 256) prow[i] = wpe[(long)(t + 1) * D_E + i];
  __syncthreads();
  float sc[2];
#pragma unroll
  for (int j = 0; j < 2; ++j) {
    const int c = tid + j * 256;
    const float* pc = wpe + (long)c * D_E;
    float dot = 0.f;
    for (int d = 0; d < D_E; ++d) dot += prow[d] * pc[d];
    dot *= 0.044194173824159216f;  // 1/sqrt(512)
    dot = fminf(fmaxf(dot, -10.f), 10.f);
    sc[j] = (c <= t) ? dot : -3.0e38f;
  }
  const float mx = block_red256(fmaxf(sc[0], sc[1]), red, true);
  const float e0 = (tid <= t) ? __expf(sc[0] - mx) : 0.f;
  const float e1 = (tid + 256 <= t) ? __expf(sc[1] - mx) : 0.f;
  const float ssum = block_red256(e0 + e1, red, false);
  const float scale = 1.0f / (ssum * (float)(t + 1));
  krn_bf[(long)t * SEQ + tid] = (__bf16)(e0 * scale);
  krn_bf[(long)t * SEQ + tid + 256] = (__bf16)(e1 * scale);
}

// ---------------- per-layer: vocab softmax stats (over s) ----------------
// L[v,s] = wte[v]·f_k[b,s]; m[b,v]=max_s L; z=sum_s exp(L-m).
// The wave's 16 vocab rows (full K=512) stay RESIDENT in 128 VGPRs (afrag),
// so the s-loop does only B-fragment loads + WMMA: A is loaded exactly once.
// Per-lane online max/sumexp; one 16-lane merge at the end.
__global__ __launch_bounds__(256, 1) void k_pass1(
    const __bf16* __restrict__ wte_bf, const __bf16* __restrict__ fkbf,
    float* __restrict__ mbuf, float* __restrict__ rzbuf) {
  const int b = blockIdx.y;
  const int wave = threadIdx.x >> 5;
  const int lane = threadIdx.x & 31;
  const int vr = blockIdx.x * 128 + wave * 16;  // 16 vocab rows per wave
  const __bf16* fk = fkbf + (long)b * SEQ * D_E;
  v16bf afrag[16];
#pragma unroll
  for (int kk = 0; kk < 16; ++kk) afrag[kk] = ld_A(wte_bf, D_E, vr, kk * 32);
  float m[8], z[8];
#pragma unroll
  for (int r = 0; r < 8; ++r) { m[r] = -3.0e38f; z[r] = 0.f; }
  for (int s0 = 0; s0 < SEQ; s0 += 32) {
    v8f acc0 = zero8(), acc1 = zero8();
#pragma unroll
    for (int kk = 0; kk < 16; ++kk) {
      v16bf b0 = ld_BT(fk, D_E, kk * 32, s0);
      v16bf b1 = ld_BT(fk, D_E, kk * 32, s0 + 16);
      acc0 = wmma_bf16(afrag[kk], b0, acc0);
      acc1 = wmma_bf16(afrag[kk], b1, acc1);
    }
    // per-lane online softmax update (lane owns cols s0+n and s0+16+n)
#pragma unroll
    for (int r = 0; r < 8; ++r) {
      const float x0 = acc0[r], x1 = acc1[r];
      const float mn = fmaxf(m[r], fmaxf(x0, x1));
      z[r] = z[r] * __expf(m[r] - mn) + __expf(x0 - mn) + __expf(x1 - mn);
      m[r] = mn;
    }
  }
  // merge the 16 lanes of each half (rows 0-7 in lanes 0-15, rows 8-15 above)
#pragma unroll
  for (int r = 0; r < 8; ++r) {
    float mall = m[r];
#pragma unroll
    for (int off = 1; off < 16; off <<= 1) mall = fmaxf(mall, __shfl_xor(mall, off, 32));
    float zz = z[r] * __expf(m[r] - mall);
#pragma unroll
    for (int off = 1; off < 16; off <<= 1) zz += __shfl_xor(zz, off, 32);
    m[r] = mall;
    z[r] = zz;
  }
  if ((lane & 15) == 0) {
    const int mh = (lane >> 4) << 3;
#pragma unroll
    for (int r = 0; r < 8; ++r) {
      const long vg = (long)b * VOC + vr + mh + r;
      mbuf[vg] = m[r];
      rzbuf[vg] = 1.0f / z[r];
    }
  }
}

// ---------------- per-layer: num/den accumulation over vocab ----------------
// Block owns (b, 16-row s-tile, v-split); num[s,0..511] kept in WMMA accumulators.
// f_k B-fragments are hoisted out of the v-loop; wte columns come from the
// transposed copy so every fragment load is a contiguous 32B per lane.
__global__ __launch_bounds__(256, 1) void k_pass2(
    const __bf16* __restrict__ wte_bf, const __bf16* __restrict__ wteT_bf,
    const __bf16* __restrict__ fkbf,
    const float* __restrict__ mbuf, const float* __restrict__ rzbuf,
    float* __restrict__ nump, float* __restrict__ denp) {
  __shared__ float Ppart[8][32][16];   // per-wave K-split partials of L-tile
  __shared__ float Pexp[32][16];       // softmaxed weights (f32, for den)
  __shared__ __bf16 Astage[16][34];    // P^T staged in A-fragment layout
  const int s0 = blockIdx.x * 16;
  const int b = blockIdx.y;
  const int split = blockIdx.z;
  const int tid = threadIdx.x;
  const int wave = tid >> 5;
  const int lane = tid & 31;
  const __bf16* fk = fkbf + (long)b * SEQ * D_E;
  v8f nacc[4];
#pragma unroll
  for (int i = 0; i < 4; ++i) nacc[i] = zero8();
  float dreg = 0.f;
  // loop-invariant f_k fragments: this wave covers k in [wave*64, wave*64+64)
  const v16bf fkA = ld_BT(fk, D_E, wave * 64, s0);
  const v16bf fkB = ld_BT(fk, D_E, wave * 64 + 32, s0);
  const long vbeg = (long)split * (VOC / NSPLIT);
  const long vend = vbeg + (VOC / NSPLIT);
  for (long v0 = vbeg; v0 < vend; v0 += 32) {
    if (v0 + 32 < vend) __builtin_prefetch(wte_bf + (v0 + 32) * D_E, 0, 1);
    // P partial GEMM: all loads issued, then 4 back-to-back WMMAs
    {
      v16bf a00 = ld_A(wte_bf, D_E, (int)v0, wave * 64);
      v16bf a10 = ld_A(wte_bf, D_E, (int)v0 + 16, wave * 64);
      v16bf a01 = ld_A(wte_bf, D_E, (int)v0, wave * 64 + 32);
      v16bf a11 = ld_A(wte_bf, D_E, (int)v0 + 16, wave * 64 + 32);
      v8f p0 = zero8(), p1 = zero8();
      p0 = wmma_bf16(a00, fkA, p0);
      p1 = wmma_bf16(a10, fkA, p1);
      p0 = wmma_bf16(a01, fkB, p0);
      p1 = wmma_bf16(a11, fkB, p1);
      const int col = lane & 15, mh = (lane >> 4) << 3;
#pragma unroll
      for (int r = 0; r < 8; ++r) {
        Ppart[wave][mh + r][col] = p0[r];
        Ppart[wave][16 + mh + r][col] = p1[r];
      }
    }
    __syncthreads();
    // deterministic cross-wave reduction + softmax weight
#pragma unroll
    for (int it = 0; it < 2; ++it) {
      const int i = tid + it * 256;
      const int vv = i >> 4, ss = i & 15;
      float sum = 0.f;
#pragma unroll
      for (int w = 0; w < 8; ++w) sum += Ppart[w][vv][ss];
      const long vg = (long)b * VOC + v0 + vv;
      const float pe = __expf(sum - mbuf[vg]) * rzbuf[vg];
      Pexp[vv][ss] = pe;
      Astage[ss][vv] = (__bf16)pe;  // transpose: A'[s][v]
    }
    __syncthreads();
    if (tid < 16) {
#pragma unroll
      for (int vv = 0; vv < 32; ++vv) dreg += Pexp[vv][tid];
    }
    // num[s, d] += sum_{v<32} Pexp[v][s] * wte[v0+v][d]; wave owns 64 d-columns
    v16bf afr = ld_A(&Astage[0][0], 34, 0, 0);
#pragma unroll
    for (int dt = 0; dt < 4; ++dt) {
      v16bf bfr = ld_BT(wteT_bf, VOC, (int)v0, wave * 64 + dt * 16);
      nacc[dt] = wmma_bf16(afr, bfr, nacc[dt]);
    }
    __syncthreads();  // before LDS reuse next iteration
  }
  float* np = nump + ((long)split * BSZ + b) * SEQ * D_E;
  const int col = lane & 15, mh = (lane >> 4) << 3;
#pragma unroll
  for (int dt = 0; dt < 4; ++dt) {
    const int d0 = wave * 64 + dt * 16;
#pragma unroll
    for (int r = 0; r < 8; ++r)
      np[(long)(s0 + mh + r) * D_E + d0 + col] = nacc[dt][r];
  }
  if (tid < 16) denp[((long)split * BSZ + b) * SEQ + s0 + tid] = dreg;
}

// V = e - num/den  (reduce the NSPLIT partials), emit bf16
__global__ void k_finishV(const float* __restrict__ nump, const float* __restrict__ denp,
                          const float* __restrict__ e, __bf16* __restrict__ vbf) {
  const long i = (long)blockIdx.x * 256 + threadIdx.x;  // over B*S*D
  const long bs = i >> 9;
  float num = 0.f, den = 0.f;
#pragma unroll
  for (int sp = 0; sp < NSPLIT; ++sp) {
    num += nump[(long)sp * ((long)BSZ * SEQ * D_E) + i];
    den += denp[(long)sp * (BSZ * SEQ) + bs];
  }
  vbf[i] = (__bf16)(e[i] - num / den);
}

// delta1 = krn2 @ V   (1/(t+1) already folded into krn2)
__global__ void k_delta(const __bf16* __restrict__ krn_bf, const __bf16* __restrict__ vbf,
                        __bf16* __restrict__ d1bf) {
  const int t0 = blockIdx.x * 16;
  const int b = blockIdx.z;
  const int wave = threadIdx.x >> 5, lane = threadIdx.x & 31;
  const int d0 = blockIdx.y * 128 + wave * 16;
  const __bf16* Vb = vbf + (long)b * SEQ * D_E;
  v8f acc = zero8();
  for (int k0 = 0; k0 < SEQ; k0 += 64) {
    v16bf aA = ld_A(krn_bf, SEQ, t0, k0);
    v16bf bA = ld_B(Vb, D_E, k0, d0);
    v16bf aB = ld_A(krn_bf, SEQ, t0, k0 + 32);
    v16bf bB = ld_B(Vb, D_E, k0 + 32, d0);
    acc = wmma_bf16(aA, bA, acc);
    acc = wmma_bf16(aB, bB, acc);
  }
  const int col = lane & 15, mh = (lane >> 4) << 3;
  __bf16* outp = d1bf + (long)b * SEQ * D_E;
#pragma unroll
  for (int r = 0; r < 8; ++r)
    outp[(long)(t0 + mh + r) * D_E + d0 + col] = (__bf16)acc[r];
}

// f_k += delta1 @ wosum^T
__global__ void k_wo(const __bf16* __restrict__ d1bf, const __bf16* __restrict__ wos_bf,
                     float* __restrict__ fk) {
  const int t0 = blockIdx.x * 16;
  const int b = blockIdx.z;
  const int wave = threadIdx.x >> 5, lane = threadIdx.x & 31;
  const int i0 = blockIdx.y * 128 + wave * 16;
  const __bf16* A = d1bf + (long)b * SEQ * D_E;
  v8f acc = zero8();
  for (int k0 = 0; k0 < D_E; k0 += 64) {
    v16bf aA = ld_A(A, D_E, t0, k0);
    v16bf bA = ld_BT(wos_bf, D_E, k0, i0);
    v16bf aB = ld_A(A, D_E, t0, k0 + 32);
    v16bf bB = ld_BT(wos_bf, D_E, k0 + 32, i0);
    acc = wmma_bf16(aA, bA, acc);
    acc = wmma_bf16(aB, bB, acc);
  }
  const int col = lane & 15, mh = (lane >> 4) << 3;
  float* outp = fk + (long)b * SEQ * D_E;
#pragma unroll
  for (int r = 0; r < 8; ++r) {
    const long idx = (long)(t0 + mh + r) * D_E + i0 + col;
    outp[idx] += acc[r];
  }
}

// h = layernorm(f_k) * w  -> bf16
__global__ void k_ln(const float* __restrict__ fk, const float* __restrict__ w,
                     __bf16* __restrict__ hbf) {
  __shared__ float red[8];
  const long row = blockIdx.x;  // b*S + s
  const float* f = fk + row * D_E;
  const int t = threadIdx.x;
  const float x0 = f[t], x1 = f[t + 256];
  const float mean = block_red256(x0 + x1, red, false) * (1.0f / D_E);
  const float a0 = x0 - mean, a1 = x1 - mean;
  const float var = block_red256(a0 * a0 + a1 * a1, red, false) * (1.0f / D_E);
  const float rs = rsqrtf(var + 1e-5f);
  hbf[row * D_E + t] = (__bf16)(a0 * rs * w[t]);
  hbf[row * D_E + t + 256] = (__bf16)(a1 * rs * w[t + 256]);
}

// g = gelu(h @ w1^T) -> bf16
__global__ void k_mlp1(const __bf16* __restrict__ hbf, const __bf16* __restrict__ w1bf,
                       __bf16* __restrict__ gbf) {
  const int t0 = blockIdx.x * 16;
  const int b = blockIdx.z;
  const int wave = threadIdx.x >> 5, lane = threadIdx.x & 31;
  const int f0 = blockIdx.y * 128 + wave * 16;
  const __bf16* A = hbf + (long)b * SEQ * D_E;
  v8f acc = zero8();
  for (int k0 = 0; k0 < D_E; k0 += 64) {
    v16bf aA = ld_A(A, D_E, t0, k0);
    v16bf bA = ld_BT(w1bf, D_E, k0, f0);
    v16bf aB = ld_A(A, D_E, t0, k0 + 32);
    v16bf bB = ld_BT(w1bf, D_E, k0 + 32, f0);
    acc = wmma_bf16(aA, bA, acc);
    acc = wmma_bf16(aB, bB, acc);
  }
  const int col = lane & 15, mh = (lane >> 4) << 3;
  __bf16* outp = gbf + (long)b * SEQ * DFF;
#pragma unroll
  for (int r = 0; r < 8; ++r) {
    const float x = acc[r];
    const float g = 0.5f * x * (1.0f + erff(x * 0.7071067811865475f));  // exact gelu
    outp[(long)(t0 + mh + r) * DFF + f0 + col] = (__bf16)g;
  }
}

// f_k += g @ w2^T ; refresh bf16 copy of f_k for the next layer
__global__ void k_mlp2(const __bf16* __restrict__ gbf, const __bf16* __restrict__ w2bf,
                       float* __restrict__ fk, __bf16* __restrict__ fkbf) {
  const int t0 = blockIdx.x * 16;
  const int b = blockIdx.z;
  const int wave = threadIdx.x >> 5, lane = threadIdx.x & 31;
  const int d0 = blockIdx.y * 128 + wave * 16;
  const __bf16* A = gbf + (long)b * SEQ * DFF;
  v8f acc = zero8();
  for (int k0 = 0; k0 < DFF; k0 += 64) {
    v16bf aA = ld_A(A, DFF, t0, k0);
    v16bf bA = ld_BT(w2bf, DFF, k0, d0);
    v16bf aB = ld_A(A, DFF, t0, k0 + 32);
    v16bf bB = ld_BT(w2bf, DFF, k0 + 32, d0);
    acc = wmma_bf16(aA, bA, acc);
    acc = wmma_bf16(aB, bB, acc);
  }
  const int col = lane & 15, mh = (lane >> 4) << 3;
  const long base = (long)b * SEQ * D_E;
#pragma unroll
  for (int r = 0; r < 8; ++r) {
    const long idx = base + (long)(t0 + mh + r) * D_E + d0 + col;
    const float nf = fk[idx] + acc[r];
    fk[idx] = nf;
    fkbf[idx] = (__bf16)nf;
  }
}

// final layernorm of last position only -> xn[b][D] (f32)
__global__ void k_lnf(const float* __restrict__ fk, const float* __restrict__ w,
                      float* __restrict__ xn) {
  __shared__ float red[8];
  const int b = blockIdx.x;
  const float* f = fk + ((long)b * SEQ + (SEQ - 1)) * D_E;
  const int t = threadIdx.x;
  const float x0 = f[t], x1 = f[t + 256];
  const float mean = block_red256(x0 + x1, red, false) * (1.0f / D_E);
  const float a0 = x0 - mean, a1 = x1 - mean;
  const float var = block_red256(a0 * a0 + a1 * a1, red, false) * (1.0f / D_E);
  const float rs = rsqrtf(var + 1e-5f);
  xn[(long)b * D_E + t] = a0 * rs * w[t];
  xn[(long)b * D_E + t + 256] = a1 * rs * w[t + 256];
}

// logits[b, v] = xn[b]·wte[v]  (f32 for accuracy; only 2x32000 rows)
__global__ void k_logits(const float* __restrict__ xn, const float* __restrict__ wte,
                         float* __restrict__ out) {
  __shared__ float xr[D_E];
  const int b = blockIdx.y;
  const long v = (long)blockIdx.x * 128 + threadIdx.x;
  for (int i = threadIdx.x; i < D_E; i += 128) xr[i] = xn[(long)b * D_E + i];
  __syncthreads();
  const float* wv = wte + v * D_E;
  float s = 0.f;
  for (int d = 0; d < D_E; ++d) s += xr[d] * wv[d];
  out[(long)b * VOC + v] = s;
}

extern "C" void kernel_launch(void* const* d_in, const int* in_sizes, int n_in,
                              void* d_out, int out_size, void* d_ws, size_t ws_size,
                              hipStream_t stream) {
  (void)in_sizes; (void)n_in; (void)out_size; (void)ws_size;
  const int*   x    = (const int*)  d_in[0];
  const float* wte  = (const float*)d_in[1];
  const float* wpe  = (const float*)d_in[2];
  const float* w_o  = (const float*)d_in[3];
  const float* w1   = (const float*)d_in[4];
  const float* w2   = (const float*)d_in[5];
  const float* lnmw = (const float*)d_in[6];
  const float* lnfw = (const float*)d_in[7];
  float* out = (float*)d_out;

  // deterministic workspace carve-out (~91 MB)
  char* wp = (char*)d_ws;
  auto take = [&](size_t bytes) {
    char* p = wp;
    wp += (bytes + 255) & ~(size_t)255;
    return p;
  };
  __bf16* wte_bf  = (__bf16*)take((size_t)VOC * D_E * 2);
  __bf16* wteT_bf = (__bf16*)take((size_t)D_E * VOC * 2);
  __bf16* w1_bf   = (__bf16*)take((size_t)DFF * D_E * 2);
  __bf16* w2_bf   = (__bf16*)take((size_t)D_E * DFF * 2);
  __bf16* wos_bf  = (__bf16*)take((size_t)D_E * D_E * 2);
  __bf16* krn_bf  = (__bf16*)take((size_t)SEQ * SEQ * 2);
  float*  ebuf    = (float*) take((size_t)BSZ * SEQ * D_E * 4);
  float*  fk      = (float*) take((size_t)BSZ * SEQ * D_E * 4);
  __bf16* fkbf    = (__bf16*)take((size_t)BSZ * SEQ * D_E * 2);
  float*  mbuf    = (float*) take((size_t)BSZ * VOC * 4);
  float*  rzbuf   = (float*) take((size_t)BSZ * VOC * 4);
  float*  nump    = (float*) take((size_t)NSPLIT * BSZ * SEQ * D_E * 4);
  float*  denp    = (float*) take((size_t)NSPLIT * BSZ * SEQ * 4);
  __bf16* vbf     = (__bf16*)take((size_t)BSZ * SEQ * D_E * 2);
  __bf16* d1bf    = (__bf16*)take((size_t)BSZ * SEQ * D_E * 2);
  __bf16* hbf     = (__bf16*)take((size_t)BSZ * SEQ * D_E * 2);
  __bf16* gbf     = (__bf16*)take((size_t)BSZ * SEQ * DFF * 2);
  float*  xnb     = (float*) take((size_t)BSZ * D_E * 4);

  k_cvt<<<(VOC * D_E) / 256, 256, 0, stream>>>(wte, wte_bf, (long)VOC * D_E);
  k_transpose<<<dim3(VOC / 64, D_E / 64), 256, 0, stream>>>(wte, wteT_bf);
  k_cvt<<<(DFF * D_E) / 256, 256, 0, stream>>>(w1, w1_bf, (long)DFF * D_E);
  k_cvt<<<(D_E * DFF) / 256, 256, 0, stream>>>(w2, w2_bf, (long)D_E * DFF);
  k_wosum<<<(D_E * D_E) / 256, 256, 0, stream>>>(w_o, wos_bf);
  k_krn<<<SEQ, 256, 0, stream>>>(wpe, krn_bf);
  k_embed<<<(BSZ * SEQ * D_E) / 256, 256, 0, stream>>>(x, wte, ebuf, fk, fkbf);

  for (int l = 0; l < NLAYER; ++l) {
    k_pass1<<<dim3(VOC / 128, BSZ), 256, 0, stream>>>(wte_bf, fkbf, mbuf, rzbuf);
    k_pass2<<<dim3(SEQ / 16, BSZ, NSPLIT), 256, 0, stream>>>(wte_bf, wteT_bf, fkbf,
                                                             mbuf, rzbuf, nump, denp);
    k_finishV<<<(BSZ * SEQ * D_E) / 256, 256, 0, stream>>>(nump, denp, ebuf, vbf);
    k_delta<<<dim3(SEQ / 16, D_E / 128, BSZ), 256, 0, stream>>>(krn_bf, vbf, d1bf);
    k_wo<<<dim3(SEQ / 16, D_E / 128, BSZ), 256, 0, stream>>>(d1bf, wos_bf, fk);
    k_ln<<<BSZ * SEQ, 256, 0, stream>>>(fk, lnmw, hbf);
    k_mlp1<<<dim3(SEQ / 16, DFF / 128, BSZ), 256, 0, stream>>>(hbf, w1_bf, gbf);
    k_mlp2<<<dim3(SEQ / 16, D_E / 128, BSZ), 256, 0, stream>>>(gbf, w2_bf, fk, fkbf);
  }

  k_lnf<<<BSZ, 256, 0, stream>>>(fk, lnfw, xnb);
  k_logits<<<dim3(VOC / 128, BSZ), 128, 0, stream>>>(xnb, wte, out);
}
